// Seq2SeqAttentionDecoder_3272765079943
// MI455X (gfx1250) — matrix operand order repaired and checked by reading, not compiled, not run on hardware
//
#include <hip/hip_runtime.h>
#include <hip/hip_bf16.h>

// Problem constants (match reference)
constexpr int B = 64, T = 64, S = 512, E = 256, H = 1024, C = 256;
constexpr int H3 = 3 * H;

typedef __attribute__((ext_vector_type(16))) _Float16 v16h;
typedef __attribute__((ext_vector_type(8)))  _Float16 v8h;
typedef __attribute__((ext_vector_type(8)))  float    v8f;

// ---------------------------------------------------------------------------
// Elementwise converts / init
// ---------------------------------------------------------------------------
__global__ void cvt_f32_f16_kernel(const float* __restrict__ src,
                                   _Float16* __restrict__ dst, int n) {
  int i = blockIdx.x * blockDim.x + threadIdx.x;
  if (i < n) dst[i] = (_Float16)src[i];
}

__global__ void init_h_kernel(const float* __restrict__ src,
                              float* __restrict__ d32,
                              _Float16* __restrict__ d16, int n) {
  int i = blockIdx.x * blockDim.x + threadIdx.x;
  if (i < n) {
    float v = src[i];
    d32[i] = v;
    d16[i] = (_Float16)v;
  }
}

// ---------------------------------------------------------------------------
// WMMA GEMM:  Cout[M,N] = A[M,K](f16) @ W[N,K](f16)^T  (+ bias[N]) (+ Add[M,N])
// A row-major (lda); W row-major out_features x in_features (ldw, sliceable).
// Block = 128 threads (4 wave32); each wave computes a 16(M) x 64(N) tile:
// 1 A fragment vs 4 B fragments per K-step of 32 -> 4 v_wmma per step.
// K-loop is software-pipelined: fragments for step i+1 are loaded before the
// WMMAs of step i issue, so WMMAs wait on loads a full iteration old
// (critical for the latency-bound M=64 recurrent GEMMs).
// M, N multiples of 64; K multiple of 32 (true at all call sites).
// ---------------------------------------------------------------------------
template <typename OutT>
__global__ void wmma_gemm_kernel(const _Float16* __restrict__ A, int lda,
                                 const _Float16* __restrict__ W, int ldw,
                                 OutT* __restrict__ Cout, int ldc,
                                 const float* __restrict__ bias,
                                 const float* __restrict__ addsrc, long long ldadd,
                                 int M, int N, int K) {
  const int wave = threadIdx.x >> 5;
  const int lane = threadIdx.x & 31;
  const int half = lane >> 4;        // 0: lanes 0-15, 1: lanes 16-31
  const int l15  = lane & 15;

  const int tileM = blockIdx.y * 64 + wave * 16;
  const int tileN = blockIdx.x * 64;
  if (tileM >= M) return;            // wave-uniform (M multiple of 16)

  const _Float16* Arow = A + (long long)(tileM + l15) * lda;
  const _Float16* Wrow[4];
#pragma unroll
  for (int j = 0; j < 4; ++j)
    Wrow[j] = W + (long long)(tileN + j * 16 + l15) * ldw;

  // ---- fragment loaders (ISA 7.12.2 16-bit layouts) ----
  // A 16x32: lanes 0-15: K = k0+{0..7,16..23}; lanes 16-31: K = k0+{8..15,24..31}
  auto loadA = [&](int k0) {
    const int ka = k0 + half * 8;
    v8h lo = *(const v8h*)(Arow + ka);
    v8h hi = *(const v8h*)(Arow + ka + 16);
    v16h f;
#pragma unroll
    for (int i = 0; i < 8; ++i) { f[i] = lo[i]; f[8 + i] = hi[i]; }
    return f;
  };
  // B 32x16: lane holds column n = l15; K = kb+{0..7,16..23} in W row n (same order as A)
  auto loadB = [&](int k0, int j) {
    const int kb = k0 + half * 8;
    v8h lo = *(const v8h*)(Wrow[j] + kb);
    v8h hi = *(const v8h*)(Wrow[j] + kb + 16);
    v16h f;
#pragma unroll
    for (int i = 0; i < 8; ++i) { f[i] = lo[i]; f[8 + i] = hi[i]; }
    return f;
  };

  v8f acc[4] = {};

  // ---- software-pipelined K loop (double-buffered fragments) ----
  v16h aCur = loadA(0);
  v16h bCur[4];
#pragma unroll
  for (int j = 0; j < 4; ++j) bCur[j] = loadB(0, j);

  for (int k0 = 32; k0 < K; k0 += 32) {
    v16h aNxt = loadA(k0);           // issue next-step loads first
    v16h bNxt[4];
#pragma unroll
    for (int j = 0; j < 4; ++j) bNxt[j] = loadB(k0, j);
#pragma unroll
    for (int j = 0; j < 4; ++j)      // compute on fragments loaded last iter
      acc[j] = __builtin_amdgcn_wmma_f32_16x16x32_f16(
          false, aCur, false, bCur[j], (short)0, acc[j], false, false);
    aCur = aNxt;
#pragma unroll
    for (int j = 0; j < 4; ++j) bCur[j] = bNxt[j];
  }
#pragma unroll
  for (int j = 0; j < 4; ++j)
    acc[j] = __builtin_amdgcn_wmma_f32_16x16x32_f16(
        false, aCur, false, bCur[j], (short)0, acc[j], false, false);

  // ---- Epilogue: C/D 16x16 f32 layout: VGPR r -> row (r + half*8), col l15
#pragma unroll
  for (int j = 0; j < 4; ++j) {
    const int ncol = tileN + j * 16 + l15;
    const float bv = bias ? bias[ncol] : 0.0f;
#pragma unroll
    for (int r = 0; r < 8; ++r) {
      const int mrow = tileM + r + half * 8;
      float v = acc[j][r] + bv;
      if (addsrc) v += addsrc[(long long)mrow * ldadd + ncol];
      Cout[(long long)mrow * ldc + ncol] = (OutT)v;
    }
  }
}

// ---------------------------------------------------------------------------
// Fused attention: per batch b
//   scores[s] = sum_h tanh(q[b,h] + keys16[b,s,h]) * wv[h]
//   attn      = softmax(scores)
//   ctx[b,h]  = sum_s attn[s] * enc16[b,s,h]   -> written f16 (GRU GEMM input)
// Block = 512 threads (16 wave32), grid = B.
// keys16+enc16 together are 134 MB f16 -> resident in the 192 MB L2 across
// all 64 decode steps (the point of the f16 staging).
// ---------------------------------------------------------------------------
__global__ void attention_kernel(const float* __restrict__ q,       // (B,H) f32
                                 const _Float16* __restrict__ keys, // (B,S,H)
                                 const _Float16* __restrict__ enc,  // (B,S,H)
                                 const float* __restrict__ wv,      // (H)
                                 _Float16* __restrict__ xctx) {     // (B,H) f16
  __shared__ float qs[H];
  __shared__ float wvs[H];
  __shared__ float sc[S];
  __shared__ float red[512];

  const int b    = blockIdx.x;
  const int tid  = threadIdx.x;
  const int wave = tid >> 5;
  const int lane = tid & 31;

  for (int i = tid; i < H; i += 512) {
    qs[i]  = q[b * H + i];
    wvs[i] = wv[i];
  }
  __syncthreads();

  // scores: each wave reduces one s at a time (lanes split H, coalesced)
  const _Float16* kb = keys + (long long)b * S * H;
  for (int si = 0; si < S / 16; ++si) {
    const int s = si * 16 + wave;
    const _Float16* krow = kb + (long long)s * H;
    float acc = 0.0f;
    for (int h = lane; h < H; h += 32)
      acc += tanhf(qs[h] + (float)krow[h]) * wvs[h];
#pragma unroll
    for (int m = 16; m >= 1; m >>= 1)
      acc += __shfl_xor(acc, m, 32);
    if (lane == 0) sc[s] = acc;
  }
  __syncthreads();

  // softmax over S=512 (one score per thread)
  float my = sc[tid];
  red[tid] = my;
  __syncthreads();
  for (int st = 256; st > 0; st >>= 1) {
    if (tid < st) red[tid] = fmaxf(red[tid], red[tid + st]);
    __syncthreads();
  }
  const float mx = red[0];
  __syncthreads();
  const float e = expf(my - mx);
  red[tid] = e;
  __syncthreads();
  for (int st = 256; st > 0; st >>= 1) {
    if (tid < st) red[tid] += red[tid + st];
    __syncthreads();
  }
  const float inv = 1.0f / red[0];
  __syncthreads();
  sc[tid] = e * inv;
  __syncthreads();

  // context: thread owns columns h=tid and h=tid+512 (coalesced over s)
  const _Float16* eb = enc + (long long)b * S * H;
  float a0 = 0.0f, a1 = 0.0f;
  for (int s = 0; s < S; ++s) {
    const float a = sc[s];
    const _Float16* erow = eb + (long long)s * H;
    a0 += a * (float)erow[tid];
    a1 += a * (float)erow[tid + 512];
  }
  xctx[b * H + tid]       = (_Float16)a0;
  xctx[b * H + tid + 512] = (_Float16)a1;
}

// ---------------------------------------------------------------------------
// GRU elementwise update (PyTorch gate order r,z,n). In-place safe on h.
// Optionally records h_new (f16) into outs(B,T,H) at column t.
// ---------------------------------------------------------------------------
__global__ void gru_update_kernel(const float* __restrict__ gi,
                                  const float* __restrict__ gh,
                                  const float* __restrict__ h,
                                  float* __restrict__ hnew,
                                  _Float16* __restrict__ hnew16,
                                  _Float16* __restrict__ outs16, int t) {
  const int idx = blockIdx.x * blockDim.x + threadIdx.x;
  if (idx >= B * H) return;
  const int b = idx / H;
  const int hh = idx - b * H;
  const long long g = (long long)b * H3 + hh;
  const float r = 1.0f / (1.0f + expf(-(gi[g] + gh[g])));
  const float z = 1.0f / (1.0f + expf(-(gi[g + H] + gh[g + H])));
  const float n = tanhf(gi[g + 2 * H] + r * gh[g + 2 * H]);
  const float hv = (1.0f - z) * n + z * h[idx];
  hnew[idx] = hv;
  hnew16[idx] = (_Float16)hv;
  if (outs16) outs16[((long long)b * T + t) * H + hh] = (_Float16)hv;
}

// ---------------------------------------------------------------------------
// Host
// ---------------------------------------------------------------------------
static inline int cdiv(int a, int b) { return (a + b - 1) / b; }

extern "C" void kernel_launch(void* const* d_in, const int* in_sizes, int n_in,
                              void* d_out, int out_size, void* d_ws, size_t ws_size,
                              hipStream_t stream) {
  const float* X    = (const float*)d_in[0];   // (B,T,E)
  const float* enc  = (const float*)d_in[1];   // (B,S,H)
  const float* h0in = (const float*)d_in[2];   // (L,B,H)
  const float* Wq   = (const float*)d_in[3];   // (H,H)
  const float* Wk   = (const float*)d_in[4];   // (H,H)
  const float* wv   = (const float*)d_in[5];   // (H)
  const float* Wih0 = (const float*)d_in[6];   // (3H, H+E)
  const float* Whh0 = (const float*)d_in[7];   // (3H, H)
  const float* bih0 = (const float*)d_in[8];
  const float* bhh0 = (const float*)d_in[9];
  const float* Wih1 = (const float*)d_in[10];  // (3H, H)
  const float* Whh1 = (const float*)d_in[11];  // (3H, H)
  const float* bih1 = (const float*)d_in[12];
  const float* bhh1 = (const float*)d_in[13];
  const float* Wd   = (const float*)d_in[14];  // (C, H)
  const float* bd   = (const float*)d_in[15];
  float* out = (float*)d_out;                  // (B,T,C)

  // ---- workspace carve-up ----
  size_t off = 0;
  auto alloc = [&](size_t bytes) -> void* {
    void* p = (char*)d_ws + off;
    off += (bytes + 255) & ~(size_t)255;
    return p;
  };
  _Float16* enc16   = (_Float16*)alloc((size_t)B * S * H * 2);
  _Float16* keys16  = (_Float16*)alloc((size_t)B * S * H * 2);
  _Float16* X16     = (_Float16*)alloc((size_t)B * T * E * 2);
  _Float16* Wq16    = (_Float16*)alloc((size_t)H * H * 2);
  _Float16* Wk16    = (_Float16*)alloc((size_t)H * H * 2);
  _Float16* Wih0_16 = (_Float16*)alloc((size_t)H3 * (H + E) * 2);
  _Float16* Whh0_16 = (_Float16*)alloc((size_t)H3 * H * 2);
  _Float16* Wih1_16 = (_Float16*)alloc((size_t)H3 * H * 2);
  _Float16* Whh1_16 = (_Float16*)alloc((size_t)H3 * H * 2);
  _Float16* Wd16    = (_Float16*)alloc((size_t)C * H * 2);
  float*    Gx      = (float*)   alloc((size_t)B * T * H3 * 4);
  float*    qbuf    = (float*)   alloc((size_t)B * H * 4);
  _Float16* xctx    = (_Float16*)alloc((size_t)B * H * 2);
  float*    h0_32   = (float*)   alloc((size_t)B * H * 4);
  _Float16* h0_16   = (_Float16*)alloc((size_t)B * H * 2);
  float*    h1_32   = (float*)   alloc((size_t)B * H * 4);
  _Float16* h1_16   = (_Float16*)alloc((size_t)B * H * 2);
  float*    gi0     = (float*)   alloc((size_t)B * H3 * 4);
  float*    gh0     = (float*)   alloc((size_t)B * H3 * 4);
  float*    gi1     = (float*)   alloc((size_t)B * H3 * 4);
  float*    gh1     = (float*)   alloc((size_t)B * H3 * 4);
  _Float16* outs16  = (_Float16*)alloc((size_t)B * T * H * 2);
  (void)ws_size; (void)in_sizes; (void)n_in; (void)out_size;

  auto cvt = [&](const float* src, _Float16* dst, int n) {
    cvt_f32_f16_kernel<<<cdiv(n, 256), 256, 0, stream>>>(src, dst, n);
  };
  auto gemm32 = [&](const _Float16* A, int lda, const _Float16* W, int ldw,
                    float* Cc, int ldc, const float* bias,
                    const float* add, long long ldadd, int M, int N, int K) {
    dim3 g(N / 64, M / 64);
    wmma_gemm_kernel<float><<<g, dim3(128), 0, stream>>>(
        A, lda, W, ldw, Cc, ldc, bias, add, ldadd, M, N, K);
  };
  auto gemm16 = [&](const _Float16* A, int lda, const _Float16* W, int ldw,
                    _Float16* Cc, int ldc, int M, int N, int K) {
    dim3 g(N / 64, M / 64);
    wmma_gemm_kernel<_Float16><<<g, dim3(128), 0, stream>>>(
        A, lda, W, ldw, Cc, ldc, nullptr, nullptr, 0, M, N, K);
  };

  // ---- one-shot setup ----
  cvt(enc,  enc16,   B * S * H);
  cvt(X,    X16,     B * T * E);
  cvt(Wq,   Wq16,    H * H);
  cvt(Wk,   Wk16,    H * H);
  cvt(Wih0, Wih0_16, H3 * (H + E));
  cvt(Whh0, Whh0_16, H3 * H);
  cvt(Wih1, Wih1_16, H3 * H);
  cvt(Whh1, Whh1_16, H3 * H);
  cvt(Wd,   Wd16,    C * H);
  init_h_kernel<<<cdiv(B * H, 256), 256, 0, stream>>>(h0in,         h0_32, h0_16, B * H);
  init_h_kernel<<<cdiv(B * H, 256), 256, 0, stream>>>(h0in + B * H, h1_32, h1_16, B * H);

  // keys_proj = enc @ Wk^T  -> f16 (stays L2-resident for all 64 steps)
  gemm16(enc16, H, Wk16, H, keys16, H, B * S, H, H);
  // Gx[b*T+t, :] = X[b,t,:] @ Wih0[:, H:H+E]^T + bih0   (hoisted x-half of gate GEMM)
  gemm32(X16, E, Wih0_16 + H, H + E, Gx, H3, bih0, nullptr, 0, B * T, H3, E);

  // ---- 64-step recurrence ----
  const int gruBlocks = cdiv(B * H, 256);
  for (int t = 0; t < T; ++t) {
    // q = h1 @ Wq^T
    gemm32(h1_16, H, Wq16, H, qbuf, H, nullptr, nullptr, 0, B, H, H);
    // attention -> xctx (f16 context)
    attention_kernel<<<B, 512, 0, stream>>>(qbuf, keys16, enc16, wv, xctx);
    // gi0 = ctx @ Wih0[:, :H]^T + Gx[:, t, :]
    gemm32(xctx, H, Wih0_16, H + E, gi0, H3, nullptr,
           Gx + (long long)t * H3, (long long)T * H3, B, H3, H);
    // gh0 = h0 @ Whh0^T + bhh0
    gemm32(h0_16, H, Whh0_16, H, gh0, H3, bhh0, nullptr, 0, B, H3, H);
    gru_update_kernel<<<gruBlocks, 256, 0, stream>>>(gi0, gh0, h0_32,
                                                     h0_32, h0_16, nullptr, t);
    // layer 1
    gemm32(h0_16, H, Wih1_16, H, gi1, H3, bih1, nullptr, 0, B, H3, H);
    gemm32(h1_16, H, Whh1_16, H, gh1, H3, bhh1, nullptr, 0, B, H3, H);
    gru_update_kernel<<<gruBlocks, 256, 0, stream>>>(gi1, gh1, h1_32,
                                                     h1_32, h1_16, outs16, t);
  }

  // outputs = outs @ Wd^T + bd   (outs stored (B,T,H) -> out is (B,T,C) directly)
  gemm32(outs16, H, Wd16, H, out, C, bd, nullptr, 0, B * T, C, H);
}
